// CrossAttention_49709951484199
// MI455X (gfx1250) — matrix operand order repaired and checked by reading, not compiled, not run on hardware
//
#include <hip/hip_runtime.h>

// ---------------------------------------------------------------------------
// CrossAttention for MI455X (gfx1250). All heavy math on v_wmma_f32_16x16x32_bf16
// (f32 accumulate). Pipeline:
//   [convert]  img,txt -> bf16 ; Wq,Wk,Wv,Wo -> transposed bf16 [N,K]
//   k = txt@Wk+bk ; v^T = (txt@Wv+bv)^T ; q = img@Wq+bq        (bf16 GEMMs)
//   S = q@k^T (f32) ; attn = softmax(S) (bf16)
//   x = attn@v ; out = x@Wo+bo (f32)
// GEMMs consume bf16 A [M,K] and bf16 B in NT layout [N,K]; tile staging uses
// DOUBLE-BUFFERED gfx1250 async copies (global_load_async_to_lds_b128, GVS
// saddr form) overlapped with WMMA, fenced by s_wait_asynccnt + barrier.
// Dims: M in {2048,8192}, N=2048, K in {1024,2048} -> no bounds checks.
// ---------------------------------------------------------------------------

typedef __attribute__((ext_vector_type(16))) __bf16         v16bf;
typedef __attribute__((ext_vector_type(8)))  __bf16         v8bf;
typedef __attribute__((ext_vector_type(8)))  float          v8f;
typedef __attribute__((ext_vector_type(8)))  unsigned short ushort8;

__device__ __forceinline__ unsigned short f32_to_bf16_rne(float x) {
    union { __bf16 h; unsigned short u; } v;
    v.h = (__bf16)x;               // native bf16 cvt on gfx1250
    return v.u;
}

// Truncated generic address of an LDS object == wave-relative LDS byte offset.
__device__ __forceinline__ unsigned lds_byte_off(const void* p) {
    return (unsigned)(unsigned long long)p;
}

// Async Global->LDS b128 copy, GVS form: saddr(SGPR u64) + voffset(VGPR i32).
// INST_OFFSET is added to the global address only (the LDS dst is passed as +16),
// so the same voffset serves consecutive 16B chunks.
__device__ __forceinline__ void async_cp_b128(unsigned lds_off, int voff,
                                              unsigned long long sbase) {
    asm volatile("global_load_async_to_lds_b128 %0, %1, %2"
                 :: "v"(lds_off), "v"(voff), "s"(sbase) : "memory");
}
__device__ __forceinline__ void async_cp_b128_o16(unsigned lds_off, int voff,
                                                  unsigned long long sbase) {
    asm volatile("global_load_async_to_lds_b128 %0, %1, %2 offset:16"
                 :: "v"(lds_off), "v"(voff), "s"(sbase) : "memory");
}
__device__ __forceinline__ void async_wait0() {
    asm volatile("s_wait_asynccnt 0x0" ::: "memory");
}

// Tile configuration -------------------------------------------------------
static constexpr int BM = 128, BN = 128, BK = 32;
static constexpr int LDT = 48;      // LDS row stride (elems): 96B, b128-aligned
static constexpr int THREADS = 256; // 8 wave32s: 4 along M x 2 along N

// C[M,N] = A[M,K](bf16) * Bt[N,K](bf16)^T (+bias).  OT: float | ushort(bf16).
// STORE_TRANS: write C transposed as [N,M] (bf16 only; 8 rows/lane contiguous).
template <typename OT, bool HAS_BIAS, bool STORE_TRANS>
__global__ __launch_bounds__(THREADS)
void gemm_bf16nt_wmma(const unsigned short* __restrict__ A,
                      const unsigned short* __restrict__ Bt,
                      const float* __restrict__ bias,
                      OT* __restrict__ C,
                      int M, int N, int K)
{
    __shared__ unsigned short As[2][BM * LDT];   // [m][k], double-buffered
    __shared__ unsigned short Bs[2][BN * LDT];   // [n][k], double-buffered

    const int t      = threadIdx.x;
    const int lane   = t & 31;
    const int laneLo = lane & 15;
    const int laneHi = lane >> 4;
    const int waveId = t >> 5;
    const int wmBase = (waveId & 3) * 32;
    const int wnBase = (waveId >> 2) * 64;

    const int rowBase = blockIdx.y * BM;
    const int colBase = blockIdx.x * BN;

    v8f acc[2][4] = {};

    // cooperative copy coords: 2 threads/row, 16 bf16 (32B = 2x b128) each
    const int ldRow = t >> 1;
    const int ldOff = (t & 1) * 16;
    const int voffA = (int)(((rowBase + ldRow) * (size_t)K + ldOff) * 2);  // bytes
    const int voffB = (int)(((colBase + ldRow) * (size_t)K + ldOff) * 2);
    const unsigned aL[2] = { lds_byte_off(&As[0][ldRow * LDT + ldOff]),
                             lds_byte_off(&As[1][ldRow * LDT + ldOff]) };
    const unsigned bL[2] = { lds_byte_off(&Bs[0][ldRow * LDT + ldOff]),
                             lds_byte_off(&Bs[1][ldRow * LDT + ldOff]) };
    const unsigned short* aG = A  + (size_t)(rowBase + ldRow) * K + ldOff;
    const unsigned short* bG = Bt + (size_t)(colBase + ldRow) * K + ldOff;

    // prologue: stage first tile into buffer 0
    {
        const unsigned long long baseA = (unsigned long long)A;
        const unsigned long long baseB = (unsigned long long)Bt;
        async_cp_b128    (aL[0],      voffA, baseA);
        async_cp_b128_o16(aL[0] + 16, voffA, baseA);
        async_cp_b128    (bL[0],      voffB, baseB);
        async_cp_b128_o16(bL[0] + 16, voffB, baseB);
        async_wait0();
    }
    __syncthreads();

    int buf = 0;
    for (int kt = 0; kt < K; kt += BK) {
        const int nxt = buf ^ 1;
        // ---- issue async copies for the NEXT tile (overlaps with WMMA) -----
        if (kt + BK < K) {
            const unsigned long long baseA = (unsigned long long)A  + (size_t)(kt + BK) * 2;
            const unsigned long long baseB = (unsigned long long)Bt + (size_t)(kt + BK) * 2;
            async_cp_b128    (aL[nxt],      voffA, baseA);
            async_cp_b128_o16(aL[nxt] + 16, voffA, baseA);
            async_cp_b128    (bL[nxt],      voffB, baseB);
            async_cp_b128_o16(bL[nxt] + 16, voffB, baseB);
            if (kt + 2 * BK < K) {                       // warm L2 two tiles out
                __builtin_prefetch(aG + kt + 2 * BK, 0, 1);
                __builtin_prefetch(bG + kt + 2 * BK, 0, 1);
            }
        }

        // ---- WMMA on the CURRENT buffer ------------------------------------
        // A 16x32 bf16 frag: lanes 0-15 K{0..7,16..23}, lanes 16-31 K{8..15,24..31}
        v16bf aFrag[2], bFrag[4];
        #pragma unroll
        for (int mi = 0; mi < 2; ++mi) {
            const unsigned short* pa =
                &As[buf][(wmBase + mi * 16 + laneLo) * LDT + laneHi * 8];
            v8bf a0 = *(const v8bf*)pa;
            v8bf a1 = *(const v8bf*)(pa + 16);
            aFrag[mi] = __builtin_shufflevector(a0, a1,
                0,1,2,3,4,5,6,7,8,9,10,11,12,13,14,15);
        }
        // B 32x16 bf16 frag: lanes 0-15 K{0..7,16..23}, lanes 16-31 K{8..15,24..31}
        #pragma unroll
        for (int ni = 0; ni < 4; ++ni) {
            const unsigned short* pb =
                &Bs[buf][(wnBase + ni * 16 + laneLo) * LDT + laneHi * 8];
            v8bf b0 = *(const v8bf*)pb;
            v8bf b1 = *(const v8bf*)(pb + 16);
            bFrag[ni] = __builtin_shufflevector(b0, b1,
                0,1,2,3,4,5,6,7,8,9,10,11,12,13,14,15);
        }
        #pragma unroll
        for (int mi = 0; mi < 2; ++mi)
            #pragma unroll
            for (int ni = 0; ni < 4; ++ni)
                acc[mi][ni] = __builtin_amdgcn_wmma_f32_16x16x32_bf16(
                    false, aFrag[mi], false, bFrag[ni],
                    (short)0, acc[mi][ni], false, false);

        // my next-tile writes done; peers' ds_loads of cur done (dscnt @ barrier)
        async_wait0();
        __syncthreads();
        buf = nxt;
    }

    // ---- epilogue: all 8 accum regs of a tile share column n = laneLo ------
    #pragma unroll
    for (int mi = 0; mi < 2; ++mi) {
        #pragma unroll
        for (int ni = 0; ni < 4; ++ni) {
            const int gcol  = colBase + wnBase + ni * 16 + laneLo;
            const int grow0 = rowBase + wmBase + mi * 16 + laneHi * 8;
            float badd = 0.0f;
            if (HAS_BIAS) badd = bias[gcol];
            if (STORE_TRANS) {
                // C is [N,M]; 8 consecutive rows -> one contiguous b128 store
                ushort8 s;
                #pragma unroll
                for (int r = 0; r < 8; ++r) s[r] = f32_to_bf16_rne(acc[mi][ni][r] + badd);
                *(ushort8*)&((unsigned short*)C)[(size_t)gcol * M + grow0] = s;
            } else {
                #pragma unroll
                for (int r = 0; r < 8; ++r) {
                    const float vv = acc[mi][ni][r] + badd;
                    if (sizeof(OT) == 4)
                        ((float*)C)[(size_t)(grow0 + r) * N + gcol] = vv;
                    else
                        ((unsigned short*)C)[(size_t)(grow0 + r) * N + gcol] = f32_to_bf16_rne(vv);
                }
            }
        }
    }
}

// ---- one-time precision/layout conversion kernels -------------------------

// f32 -> bf16, contiguous, 8 elems/thread (size must be a multiple of 2048)
__global__ __launch_bounds__(256)
void convert_f32_bf16(const float* __restrict__ in, unsigned short* __restrict__ out)
{
    const size_t i = ((size_t)blockIdx.x * 256 + threadIdx.x) * 8;
    float4 a = ((const float4*)(in + i))[0];
    float4 b = ((const float4*)(in + i))[1];
    ushort8 s;
    s[0] = f32_to_bf16_rne(a.x); s[1] = f32_to_bf16_rne(a.y);
    s[2] = f32_to_bf16_rne(a.z); s[3] = f32_to_bf16_rne(a.w);
    s[4] = f32_to_bf16_rne(b.x); s[5] = f32_to_bf16_rne(b.y);
    s[6] = f32_to_bf16_rne(b.z); s[7] = f32_to_bf16_rne(b.w);
    *(ushort8*)(out + i) = s;
}

// W[K,N] f32 -> Wt[N,K] bf16 via 32x32 LDS tile (coalesced both sides)
__global__ __launch_bounds__(256)
void transpose_f32_bf16(const float* __restrict__ W, unsigned short* __restrict__ Wt,
                        int K, int N)
{
    __shared__ float tile[32][33];
    const int k0 = blockIdx.y * 32, n0 = blockIdx.x * 32;
    const int tx = threadIdx.x & 31, ty = threadIdx.x >> 5;   // ty: 0..7
    #pragma unroll
    for (int j = 0; j < 4; ++j)
        tile[ty + j * 8][tx] = W[(size_t)(k0 + ty + j * 8) * N + n0 + tx];
    __syncthreads();
    #pragma unroll
    for (int j = 0; j < 4; ++j)
        Wt[(size_t)(n0 + ty + j * 8) * K + k0 + tx] = f32_to_bf16_rne(tile[tx][ty + j * 8]);
}

// Row softmax over N=2048 columns: one block/row, 8 elems/thread.
__global__ __launch_bounds__(256)
void softmax_rows_bf16(const float* __restrict__ S, unsigned short* __restrict__ O, int N)
{
    __shared__ float red[256];
    const int row = blockIdx.x;
    const int t   = threadIdx.x;
    const float* s = S + (size_t)row * N;

    float x[8];
    float m = -3.402823466e38f;
    #pragma unroll
    for (int j = 0; j < 8; ++j) { x[j] = s[t + j * 256]; m = fmaxf(m, x[j]); }
    red[t] = m; __syncthreads();
    for (int st = 128; st > 0; st >>= 1) {
        if (t < st) red[t] = fmaxf(red[t], red[t + st]);
        __syncthreads();
    }
    m = red[0]; __syncthreads();

    float sum = 0.0f;
    #pragma unroll
    for (int j = 0; j < 8; ++j) { x[j] = __expf(x[j] - m); sum += x[j]; }
    red[t] = sum; __syncthreads();
    for (int st = 128; st > 0; st >>= 1) {
        if (t < st) red[t] += red[t + st];
        __syncthreads();
    }
    const float inv = 1.0f / red[0];

    unsigned short* o = O + (size_t)row * N;
    #pragma unroll
    for (int j = 0; j < 8; ++j) o[t + j * 256] = f32_to_bf16_rne(x[j] * inv);
}

// ---------------------------------------------------------------------------
extern "C" void kernel_launch(void* const* d_in, const int* in_sizes, int n_in,
                              void* d_out, int out_size, void* d_ws, size_t ws_size,
                              hipStream_t stream)
{
    constexpr int N_IMG = 8192, D_IMG = 2048, N_TXT = 2048, D_TXT = 1024;

    const float* img = (const float*)d_in[0];
    const float* txt = (const float*)d_in[1];
    const float* Wq  = (const float*)d_in[2];
    const float* bq  = (const float*)d_in[3];
    const float* Wk  = (const float*)d_in[4];
    const float* bk  = (const float*)d_in[5];
    const float* Wv  = (const float*)d_in[6];
    const float* bv  = (const float*)d_in[7];
    const float* Wo  = (const float*)d_in[8];
    const float* bo  = (const float*)d_in[9];
    float* out = (float*)d_out;

    // workspace carve-out (~247 MB)
    char* ws = (char*)d_ws;
    unsigned short* img_bf = (unsigned short*)ws; ws += (size_t)N_IMG * D_IMG * 2;
    unsigned short* txt_bf = (unsigned short*)ws; ws += (size_t)N_TXT * D_TXT * 2;
    unsigned short* Wq_t   = (unsigned short*)ws; ws += (size_t)D_IMG * D_IMG * 2;
    unsigned short* Wk_t   = (unsigned short*)ws; ws += (size_t)D_IMG * D_TXT * 2;
    unsigned short* Wv_t   = (unsigned short*)ws; ws += (size_t)D_IMG * D_TXT * 2;
    unsigned short* Wo_t   = (unsigned short*)ws; ws += (size_t)D_IMG * D_IMG * 2;
    unsigned short* q_bf   = (unsigned short*)ws; ws += (size_t)N_IMG * D_IMG * 2;
    unsigned short* k_bf   = (unsigned short*)ws; ws += (size_t)N_TXT * D_IMG * 2;
    unsigned short* v_t    = (unsigned short*)ws; ws += (size_t)N_TXT * D_IMG * 2;
    float*          Smat   = (float*)ws;          ws += (size_t)N_IMG * N_TXT * 4;
    unsigned short* attn   = (unsigned short*)ws; ws += (size_t)N_IMG * N_TXT * 2;
    unsigned short* x_bf   = (unsigned short*)ws; ws += (size_t)N_IMG * D_IMG * 2;

    const dim3 blk(THREADS);

    // one-time conversions / weight transposes
    convert_f32_bf16<<<(N_IMG * D_IMG) / 2048, 256, 0, stream>>>(img, img_bf);
    convert_f32_bf16<<<(N_TXT * D_TXT) / 2048, 256, 0, stream>>>(txt, txt_bf);
    transpose_f32_bf16<<<dim3(D_IMG / 32, D_IMG / 32), 256, 0, stream>>>(Wq, Wq_t, D_IMG, D_IMG);
    transpose_f32_bf16<<<dim3(D_IMG / 32, D_TXT / 32), 256, 0, stream>>>(Wk, Wk_t, D_TXT, D_IMG);
    transpose_f32_bf16<<<dim3(D_IMG / 32, D_TXT / 32), 256, 0, stream>>>(Wv, Wv_t, D_TXT, D_IMG);
    transpose_f32_bf16<<<dim3(D_IMG / 32, D_IMG / 32), 256, 0, stream>>>(Wo, Wo_t, D_IMG, D_IMG);

    // k = txt @ Wk + bk   [N_TXT, D_IMG] bf16
    gemm_bf16nt_wmma<unsigned short, true, false>
        <<<dim3(D_IMG / BN, N_TXT / BM), blk, 0, stream>>>(txt_bf, Wk_t, bk, k_bf, N_TXT, D_IMG, D_TXT);
    // v^T = (txt @ Wv + bv)^T  -> v_t [D_IMG, N_TXT] bf16 (NT operand for attn@v)
    gemm_bf16nt_wmma<unsigned short, true, true>
        <<<dim3(D_IMG / BN, N_TXT / BM), blk, 0, stream>>>(txt_bf, Wv_t, bv, v_t, N_TXT, D_IMG, D_TXT);
    // q = img @ Wq + bq   [N_IMG, D_IMG] bf16
    gemm_bf16nt_wmma<unsigned short, true, false>
        <<<dim3(D_IMG / BN, N_IMG / BM), blk, 0, stream>>>(img_bf, Wq_t, bq, q_bf, N_IMG, D_IMG, D_IMG);
    // S = q @ k^T         [N_IMG, N_TXT] f32 (k_bf already [N,K])
    gemm_bf16nt_wmma<float, false, false>
        <<<dim3(N_TXT / BN, N_IMG / BM), blk, 0, stream>>>(q_bf, k_bf, nullptr, Smat, N_IMG, N_TXT, D_IMG);
    // attn = softmax(S, axis=1) -> bf16
    softmax_rows_bf16<<<N_IMG, 256, 0, stream>>>(Smat, attn, N_TXT);
    // x = attn @ v        [N_IMG, D_IMG] bf16 (v_t is [D_IMG, N_TXT] = NT)
    gemm_bf16nt_wmma<unsigned short, false, false>
        <<<dim3(D_IMG / BN, N_IMG / BM), blk, 0, stream>>>(attn, v_t, nullptr, x_bf, N_IMG, D_IMG, N_TXT);
    // out = x @ Wo + bo   [N_IMG, D_IMG] f32
    gemm_bf16nt_wmma<float, true, false>
        <<<dim3(D_IMG / BN, N_IMG / BM), blk, 0, stream>>>(x_bf, Wo_t, bo, out, N_IMG, D_IMG, D_IMG);
}
